// ONNXCompatibleCondConv2d_27522150432953
// MI455X (gfx1250) — matrix operand order repaired and checked by reading, not compiled, not run on hardware
//
#include <hip/hip_runtime.h>

// CondConv on MI455X (gfx1250), wave32 + WMMA.
//
// out[b] = conv2d(x[b], sum_e r[b,e] * W[e])   (conv is linear in weights)
// -> kernel 1: per-sample weight mix (f32 accum, f16 store, GEMM-ready layout)
// -> kernel 1b: x NCHW f32 -> channels-last f16 (LDS-tiled transpose)
// -> kernel 2: implicit-GEMM conv, M=COUT=128, N=56*56=3136, K=9*128=1152,
//    v_wmma_f32_16x16x32_f16, double-buffered LDS, software-pipelined staging.

#define B_SZ   16
#define CIN    128
#define COUT   128
#define H_SZ   56
#define W_SZ   56
#define E_SZ   4
#define NSPAT  (H_SZ * W_SZ)        // 3136
#define KTOT   (9 * CIN)            // 1152, ordered (kh,kw,cin)
#define NCHUNK (KTOT / 32)          // 36 K-chunks of 32
#define CHUNK_ELEMS (COUT * 32)     // 4096 halfs per (b, chunk) of Wmix

typedef __attribute__((ext_vector_type(16))) _Float16 v16h;
typedef __attribute__((ext_vector_type(8)))  float    v8f;
typedef __attribute__((ext_vector_type(4)))  int      v4i;
typedef __attribute__((ext_vector_type(2)))  int      v2i;

union Frag16 { v16h h; v4i i[2]; };

// ---------------------------------------------------------------------------
// Kernel 1: Wmix[b][c][m][kk] = f16( sum_e r[b,e] * W[e][m][cin][kh][kw] )
//   c = kpos*4 + cinBlk (kpos = kh*3+kw), cin = cinBlk*32 + kk, m = cout.
// ---------------------------------------------------------------------------
__global__ void mix_weights(const float* __restrict__ Wsrc,    // [E][COUT][CIN][3][3]
                            const float* __restrict__ rw,      // [B][E]
                            _Float16* __restrict__ Wmix) {     // [B][36][128][32]
    int gid = blockIdx.x * 256 + threadIdx.x;
    int kk  = gid & 31;
    int m   = (gid >> 5) & 127;
    int bc  = gid >> 12;
    int c   = bc % NCHUNK;
    int b   = bc / NCHUNK;
    if (b >= B_SZ) return;
    int kpos = c >> 2;
    int cin  = ((c & 3) << 5) + kk;
    float acc = 0.f;
#pragma unroll
    for (int e = 0; e < E_SZ; ++e)
        acc += rw[b * E_SZ + e] * Wsrc[((e * COUT + m) * CIN + cin) * 9 + kpos];
    Wmix[gid] = (_Float16)acc;
}

// ---------------------------------------------------------------------------
// Kernel 1b: x[b][cin][n] (f32, NCHW) -> xt[b][n][cin] (f16, channels-last).
// 32(cin) x 32(n) LDS tile; coalesced reads and writes.
// Grid: (NSPAT/32, CIN/32, B) = (98, 4, 16), block = 256.
// ---------------------------------------------------------------------------
__global__ void transpose_x(const float* __restrict__ x,
                            _Float16* __restrict__ xt) {
    __shared__ _Float16 tile[32][33];           // +1 pad to dodge bank conflicts
    const int b  = blockIdx.z;
    const int n0 = blockIdx.x * 32;
    const int c0 = blockIdx.y * 32;
    const int tn = threadIdx.x & 31;            // n within tile (read phase)
    const int tc = threadIdx.x >> 5;            // 0..7
#pragma unroll
    for (int r = 0; r < 4; ++r) {
        int cin = c0 + tc + r * 8;
        tile[tc + r * 8][tn] =
            (_Float16)x[((size_t)b * CIN + cin) * NSPAT + n0 + tn];
    }
    __syncthreads();
    const int wn = threadIdx.x >> 3;            // n within tile (write phase)
    const int wc = (threadIdx.x & 7) * 4;       // cin offset, 4 halfs per thread
    union { _Float16 h[4]; v2i i; } pk;
#pragma unroll
    for (int q = 0; q < 4; ++q) pk.h[q] = tile[wc + q][wn];
    *(v2i*)&xt[((size_t)b * NSPAT + n0 + wn) * CIN + c0 + wc] = pk.i;
}

// ---------------------------------------------------------------------------
// Kernel 2: per-sample implicit-GEMM conv.
// Block = 256 threads (8 wave32s) -> 128(M) x 128(N) output tile.
// Wave tile: 64(M) x 32(N) = 8 accumulators of v8f.
// CHLAST=true : B staged from f16 channels-last xt (2x global b128 / thread).
// CHLAST=false: fallback, B gathered from f32 NCHW x (16 strided b32 + cvt).
// ---------------------------------------------------------------------------
template <bool CHLAST>
__global__ void __launch_bounds__(256)
condconv_gemm(const float* __restrict__ x,        // [B][CIN][56][56]
              const _Float16* __restrict__ xt,    // [B][3136][128]
              const _Float16* __restrict__ Wmix,  // [B][36][128][32]
              float* __restrict__ out) {          // [B][COUT][56][56]
    __shared__ _Float16 Alds[2][CHUNK_ELEMS];   // [m=128][kk=32] per buffer
    __shared__ _Float16 Blds[2][128 * 32];      // [nl=128][kk=32] per buffer

    const int t    = threadIdx.x;
    const int b    = blockIdx.y;
    const int n0   = blockIdx.x * 128;

    const int wave = t >> 5;
    const int lane = t & 31;
    const int g    = lane >> 4;      // lane group (0: lanes 0-15, 1: 16-31)
    const int l16  = lane & 15;
    const int mW   = (wave & 1) * 64;
    const int nW   = (wave >> 1) * 32;

    // ---- B-staging per-thread invariants: one n, 16 contiguous kk ----
    const int  nl     = t & 127;
    const int  kkBase = (t >> 7) << 4;          // 0 or 16
    const int  n      = n0 + nl;
    const bool nOK    = (n < NSPAT);
    const int  h0     = n / W_SZ;
    const int  w0     = n - h0 * W_SZ;
    // invariant base pointers (per-chunk part added as a scalar offset)
    const float*    xF = x  + ((size_t)b * CIN + kkBase) * NSPAT + n;
    const _Float16* xC = xt + ((size_t)b * NSPAT + n) * CIN + kkBase;

    const v4i* srcA = (const v4i*)(Wmix + (size_t)b * NCHUNK * CHUNK_ELEMS);

    v8f acc[4][2] = {};

    // staged-chunk registers
    v4i   a0, a1;
    v4i   br0, br1;        // CHLAST path
    float bv[16];          // fallback path

    auto loadChunk = [&](int s) {
        a0 = srcA[s * 512 + t];
        a1 = srcA[s * 512 + t + 256];
        int kpos = s >> 2;
        int q3   = kpos / 3;
        int dh   = q3 - 1;
        int dw   = (kpos - q3 * 3) - 1;
        int cinBase = (s & 3) << 5;
        int h = h0 + dh, w = w0 + dw;
        bool ok  = nOK && (unsigned)h < H_SZ && (unsigned)w < W_SZ;
        int  off = dh * W_SZ + dw;               // n' = n + off
        if (CHLAST) {
            const v4i* p = (const v4i*)(xC + (ptrdiff_t)off * CIN + cinBase);
            v4i z = {};
            br0 = ok ? p[0] : z;
            br1 = ok ? p[1] : z;
        } else {
            const float* p = xF + (size_t)cinBase * NSPAT + off;
#pragma unroll
            for (int j = 0; j < 16; ++j)
                bv[j] = ok ? p[(size_t)j * NSPAT] : 0.f;
        }
    };

    auto storeChunk = [&](int buf) {
        v4i* dA = (v4i*)Alds[buf];
        dA[t]       = a0;
        dA[t + 256] = a1;
        v4i* dB = (v4i*)&Blds[buf][nl * 32 + kkBase];
        if (CHLAST) {
            dB[0] = br0;
            dB[1] = br1;
        } else {
            union { _Float16 h[16]; v4i i[2]; } pk;
#pragma unroll
            for (int j = 0; j < 16; ++j) pk.h[j] = (_Float16)bv[j];
            dB[0] = pk.i[0];
            dB[1] = pk.i[1];
        }
    };

    // prologue: stage chunk 0 into buffer 0
    loadChunk(0);
    storeChunk(0);

    for (int c = 0; c < NCHUNK; ++c) {
        __syncthreads();                         // buf[c&1] ready for all waves
        if (c + 1 < NCHUNK) loadChunk(c + 1);    // global loads overlap compute

        const _Float16* Ab = Alds[c & 1];
        const _Float16* Bb = Blds[c & 1];

        // B fragment (32x16, 16-bit): lane holds N = nBase + l16, halfs 0-7 -> K = 8g..8g+7, halfs 8-15 -> K = 16+8g..16+8g+7
        Frag16 bf[2];
#pragma unroll
        for (int nb = 0; nb < 2; ++nb) {
            const _Float16* p = &Bb[(nW + nb * 16 + l16) * 32 + g * 8];
            bf[nb].i[0] = *(const v4i*)p;
            bf[nb].i[1] = *(const v4i*)(p + 16);
        }
        // A fragment (16x32): lane holds M = mBase + l16,
        //   halfs 0-7 -> K = 8g..8g+7, halfs 8-15 -> K = 16+8g..16+8g+7
#pragma unroll
        for (int mb = 0; mb < 4; ++mb) {
            const _Float16* pa = &Ab[(mW + mb * 16 + l16) * 32 + g * 8];
            Frag16 af;
            af.i[0] = *(const v4i*)pa;
            af.i[1] = *(const v4i*)(pa + 16);
#pragma unroll
            for (int nb = 0; nb < 2; ++nb)
                acc[mb][nb] = __builtin_amdgcn_wmma_f32_16x16x32_f16(
                    false, af.h, false, bf[nb].h, (short)0, acc[mb][nb],
                    false, false);
        }

        if (c + 1 < NCHUNK) storeChunk((c + 1) & 1);  // write the other buffer
    }

    // ---- store: D layout lane l16 -> column, VGPR r -> row (r + 8g) ----
#pragma unroll
    for (int mb = 0; mb < 4; ++mb) {
#pragma unroll
        for (int nb = 0; nb < 2; ++nb) {
            int nOut = n0 + nW + nb * 16 + l16;
            if (nOut < NSPAT) {
                int mBase = mW + mb * 16 + g * 8;
#pragma unroll
                for (int r = 0; r < 8; ++r)
                    out[((size_t)b * COUT + mBase + r) * NSPAT + nOut] = acc[mb][nb][r];
            }
        }
    }
}

// ---------------------------------------------------------------------------
extern "C" void kernel_launch(void* const* d_in, const int* in_sizes, int n_in,
                              void* d_out, int out_size, void* d_ws, size_t ws_size,
                              hipStream_t stream) {
    (void)in_sizes; (void)n_in; (void)out_size;
    const float* x  = (const float*)d_in[0];   // [16][128][56][56]
    const float* rw = (const float*)d_in[1];   // [16][4]
    const float* W  = (const float*)d_in[2];   // [4][128][128][3][3]
    float* out      = (float*)d_out;           // [16][128][56][56]

    const size_t xtBytes = (size_t)B_SZ * NSPAT * CIN * sizeof(_Float16);        // 12.8 MB
    const size_t wmBytes = (size_t)B_SZ * NCHUNK * CHUNK_ELEMS * sizeof(_Float16); // 4.7 MB
    const int    total   = B_SZ * NCHUNK * CHUNK_ELEMS;                          // 2,359,296
    const dim3   grid((NSPAT + 127) / 128, B_SZ, 1);                             // 25 x 16

    if (ws_size >= xtBytes + wmBytes) {
        // fast path: channels-last f16 activations
        _Float16* xt   = (_Float16*)d_ws;
        _Float16* Wmix = (_Float16*)((char*)d_ws + xtBytes);
        mix_weights<<<total / 256, 256, 0, stream>>>(W, rw, Wmix);
        dim3 tgrid(NSPAT / 32, CIN / 32, B_SZ);                                  // 98 x 4 x 16
        transpose_x<<<tgrid, 256, 0, stream>>>(x, xt);
        condconv_gemm<true><<<grid, 256, 0, stream>>>(x, xt, Wmix, out);
    } else {
        // fallback: gather f32 NCHW directly
        _Float16* Wmix = (_Float16*)d_ws;                                        // 4.7 MB
        mix_weights<<<total / 256, 256, 0, stream>>>(W, rw, Wmix);
        condconv_gemm<false><<<grid, 256, 0, stream>>>(x, (const _Float16*)nullptr,
                                                       Wmix, out);
    }
}
